// BasisMatrixReadout_85710367359118
// MI455X (gfx1250) — matrix operand order repaired and checked by reading, not compiled, not run on hardware
//
#include <hip/hip_runtime.h>
#include <cmath>

// ---------------- problem constants ----------------
#define N_NODES   1000
#define T_TYPES   4
#define P_TYPES   10
#define D_DIM     13
#define F_DIM     128
#define H_DIM     128
#define EA_DIM    16
#define K_DIM     64
#define E_EDGES   20000
#define I_NODE_D  91
#define I_EDGE_D  169
#define ND        (N_NODES * D_DIM)          // 13000
#define FIN_EM    (2 * F_DIM + 2 * EA_DIM)   // 288
#define FIN_PROJ  (H_DIM + 2 * F_DIM)        // 384
#define INV_AVG_NEIGH 0.05f                  // 1/20

// ---------------- native vector types ----------------
typedef __attribute__((ext_vector_type(16))) __bf16 v16bf;
typedef __attribute__((ext_vector_type(8)))  __bf16 v8bf;
typedef __attribute__((ext_vector_type(8)))  float  v8f;
typedef __attribute__((ext_vector_type(4)))  float  v4f;

static __device__ __forceinline__ v8f wmma_bf16(const v16bf a, const v16bf b, const v8f c) {
    // D = A(16x32 bf16) * B(32x16 bf16) + C(16x16 f32)
    return __builtin_amdgcn_wmma_f32_16x16x32_bf16(false, a, false, b, (short)0, c, false, false);
}

// B fragment from bf16 column-major Wt[N][K]: two 16-byte loads.
// lane -> N = n0 + (lane&15) ; element e<8 -> K = k0 + (lane>>4)*8 + e, e>=8 -> K = k0 + 16 + (lane>>4)*8 + (e-8)
static __device__ __forceinline__ v16bf load_Bt(const __bf16* __restrict__ Wt, int KD,
                                                int k0, int n0, int lane) {
    const __bf16* p = Wt + (size_t)(n0 + (lane & 15)) * KD + k0 + (lane >> 4) * 8;
    v8bf lo = *(const v8bf*)p;
    v8bf hi = *(const v8bf*)(p + 16);
    v16bf b;
#pragma unroll
    for (int j = 0; j < 8; ++j) { b[j] = lo[j]; b[8 + j] = hi[j]; }
    return b;
}

static __device__ __forceinline__ v16bf pack_a(v8bf lo, v8bf hi) {
    v16bf a;
#pragma unroll
    for (int j = 0; j < 8; ++j) { a[j] = lo[j]; a[8 + j] = hi[j]; }
    return a;
}

// ---------------- zero fill ----------------
__global__ void zero_nt_f4(v4f* __restrict__ p, int n4) {
    int i = blockIdx.x * blockDim.x + threadIdx.x;
    if (i < n4) { v4f z = {0.f, 0.f, 0.f, 0.f}; __builtin_nontemporal_store(z, &p[i]); }
}
__global__ void zero_f4(v4f* __restrict__ p, int n4) {
    int i = blockIdx.x * blockDim.x + threadIdx.x;
    if (i < n4) { v4f z = {0.f, 0.f, 0.f, 0.f}; p[i] = z; }
}

// ---------------- f32 -> bf16 convert / weight transpose ----------------
__global__ void conv_bf16(const float* __restrict__ s, __bf16* __restrict__ d, int n) {
    int i = blockIdx.x * blockDim.x + threadIdx.x;
    if (i < n) d[i] = (__bf16)s[i];
}
// W[K][N] row-major f32 -> Wt[N][K] bf16
__global__ void transpose_bf16(const float* __restrict__ W, __bf16* __restrict__ Wt,
                               int K, int N) {
    int i = blockIdx.x * blockDim.x + threadIdx.x;
    if (i >= K * N) return;
    const int k = i / N, n = i % N;
    Wt[(size_t)n * K + k] = (__bf16)W[i];
}

// ---------------- GEMM 1: m = node_feats @ W_msg (1000x128x128) ----------------
// one wave per 16-row M tile; 8 N-tiles accumulated in registers
__global__ void gemm_node_msg(const __bf16* __restrict__ Xb, const __bf16* __restrict__ Wt,
                              float* __restrict__ Y) {
    const int TILES_M = (N_NODES + 15) / 16;        // 63
    const int mt = blockIdx.x * 8 + (threadIdx.x >> 5);
    if (mt >= TILES_M) return;                      // whole wave exits together
    const int lane = threadIdx.x & 31;
    const int half = lane >> 4;
    const int row = mt * 16 + (lane & 15);
    const int rowc = row < N_NODES ? row : N_NODES - 1;   // clamp: OOB C rows never stored
    const __bf16* xr = Xb + (size_t)rowc * F_DIM;

    v8f c[8];
#pragma unroll
    for (int t = 0; t < 8; ++t) c[t] = (v8f){};
#pragma unroll
    for (int k0 = 0; k0 < F_DIM; k0 += 32) {
        v8bf lo = *(const v8bf*)(xr + k0 + half * 8);
        v8bf hi = *(const v8bf*)(xr + k0 + 16 + half * 8);
        v16bf a = pack_a(lo, hi);
#pragma unroll
        for (int t = 0; t < 8; ++t)
            c[t] = wmma_bf16(a, load_Bt(Wt, F_DIM, k0, t * 16, lane), c[t]);
    }
#pragma unroll
    for (int t = 0; t < 8; ++t)
#pragma unroll
        for (int r = 0; r < 8; ++r) {
            const int orow = mt * 16 + half * 8 + r;
            if (orow < N_NODES) Y[(size_t)orow * F_DIM + t * 16 + (lane & 15)] = c[t][r];
        }
}

// ---------------- symmetric edge aggregation (atomic segment_sum) ----------------
__global__ void aggregate_kernel(const float* __restrict__ m, const int* __restrict__ src,
                                 const int* __restrict__ dst, float* __restrict__ agg) {
    int i = blockIdx.x * blockDim.x + threadIdx.x;
    if (i >= E_EDGES * F_DIM) return;
    const int e = i >> 7, f = i & 127;
    const int s = src[e], d = dst[e];
    atomicAdd(&agg[(size_t)d * F_DIM + f], m[(size_t)s * F_DIM + f]);
    atomicAdd(&agg[(size_t)s * F_DIM + f], m[(size_t)d * F_DIM + f]);
}

// ---------------- node_h = node_feats + agg/20 + node_attrs @ W_attr ----------------
__global__ void node_h_kernel(const float* __restrict__ nf, const float* __restrict__ agg,
                              const float* __restrict__ na, const float* __restrict__ W_attr,
                              float* __restrict__ nh) {
    int i = blockIdx.x * blockDim.x + threadIdx.x;
    if (i >= N_NODES * F_DIM) return;
    const int n = i >> 7, f = i & 127;
    float acc = nf[i] + agg[i] * INV_AVG_NEIGH;
#pragma unroll
    for (int t = 0; t < T_TYPES; ++t)
        acc = fmaf(na[n * T_TYPES + t], W_attr[t * F_DIM + f], acc);
    nh[i] = acc;
}

// ---------------- GEMM 2: edge_msg = tanh([nh[s], nh[d], ef, ea] @ W_em), bf16 out ----------------
__global__ void gemm_edge_msg(const __bf16* __restrict__ nhb, const __bf16* __restrict__ efb,
                              const __bf16* __restrict__ eab, const int* __restrict__ src,
                              const int* __restrict__ dst, const __bf16* __restrict__ Wt,
                              __bf16* __restrict__ Y) {
    const int mt = blockIdx.x * 8 + (threadIdx.x >> 5);
    if (mt >= E_EDGES / 16) return;
    const int lane = threadIdx.x & 31;
    const int half = lane >> 4;
    const int e = mt * 16 + (lane & 15);
    const __bf16* hs = nhb + (size_t)src[e] * F_DIM;
    const __bf16* hd = nhb + (size_t)dst[e] * F_DIM;
    const __bf16* pf = efb + (size_t)e * EA_DIM;
    const __bf16* pa = eab + (size_t)e * EA_DIM;

    auto fetch8 = [&](int k) -> v8bf {   // k is a compile-time multiple of 8 after unroll
        const __bf16* p;
        if (k < 128)      p = hs + k;
        else if (k < 256) p = hd + (k - 128);
        else if (k < 272) p = pf + (k - 256);
        else              p = pa + (k - 272);
        return *(const v8bf*)p;
    };

    v8f c[8];
#pragma unroll
    for (int t = 0; t < 8; ++t) c[t] = (v8f){};
#pragma unroll
    for (int k0 = 0; k0 < FIN_EM; k0 += 32) {
        v16bf a = pack_a(fetch8(k0 + half * 8), fetch8(k0 + 16 + half * 8));
#pragma unroll
        for (int t = 0; t < 8; ++t)
            c[t] = wmma_bf16(a, load_Bt(Wt, FIN_EM, k0, t * 16, lane), c[t]);
    }
#pragma unroll
    for (int t = 0; t < 8; ++t)
#pragma unroll
        for (int r = 0; r < 8; ++r)
            Y[(size_t)(mt * 16 + half * 8 + r) * H_DIM + t * 16 + (lane & 15)] =
                (__bf16)tanhf(c[t][r]);
}

// ---------------- GEMM 3: h = [edge_msg, nh[a], nh[b]] @ W_proj (E x 384 x 64), f32 out ----------------
__global__ void gemm_proj(const __bf16* __restrict__ emsgb, const __bf16* __restrict__ nhb,
                          const int* __restrict__ aidx, const int* __restrict__ bidx,
                          const __bf16* __restrict__ Wt, float* __restrict__ Y) {
    const int mt = blockIdx.x * 8 + (threadIdx.x >> 5);
    if (mt >= E_EDGES / 16) return;
    const int lane = threadIdx.x & 31;
    const int half = lane >> 4;
    const int e = mt * 16 + (lane & 15);
    const __bf16* pm = emsgb + (size_t)e * H_DIM;
    const __bf16* pa = nhb + (size_t)aidx[e] * F_DIM;
    const __bf16* pb = nhb + (size_t)bidx[e] * F_DIM;

    auto fetch8 = [&](int k) -> v8bf {
        const __bf16* p;
        if (k < 128)      p = pm + k;
        else if (k < 256) p = pa + (k - 128);
        else              p = pb + (k - 256);
        return *(const v8bf*)p;
    };

    v8f c[4];
#pragma unroll
    for (int t = 0; t < 4; ++t) c[t] = (v8f){};
#pragma unroll
    for (int k0 = 0; k0 < FIN_PROJ; k0 += 32) {
        v16bf a = pack_a(fetch8(k0 + half * 8), fetch8(k0 + 16 + half * 8));
#pragma unroll
        for (int t = 0; t < 4; ++t)
            c[t] = wmma_bf16(a, load_Bt(Wt, FIN_PROJ, k0, t * 16, lane), c[t]);
    }
#pragma unroll
    for (int t = 0; t < 4; ++t)
#pragma unroll
        for (int r = 0; r < 8; ++r)
            Y[(size_t)(mt * 16 + half * 8 + r) * K_DIM + t * 16 + (lane & 15)] = c[t][r];
}

// ---------------- per-edge irreps: irr[e,i] = sum_k h[e,k] * W_edge[type, k, i] ----------------
__global__ void edge_irr_kernel(const float* __restrict__ h, const int* __restrict__ et,
                                const float* __restrict__ W_edge, float* __restrict__ irr) {
    int i = blockIdx.x * blockDim.x + threadIdx.x;
    if (i >= E_EDGES * I_EDGE_D) return;
    const int e = i / I_EDGE_D, ix = i % I_EDGE_D;
    const float* hr = h + (size_t)e * K_DIM;
    const float* w = W_edge + (size_t)et[e] * K_DIM * I_EDGE_D + ix;
    float acc = 0.f;
#pragma unroll 8
    for (int k = 0; k < K_DIM; ++k) acc = fmaf(hr[k], w[(size_t)k * I_EDGE_D], acc);
    irr[i] = acc;
}

// ---------------- per-edge change of basis ----------------
__global__ void edge_cob_kernel(const float* __restrict__ irr, const int* __restrict__ et,
                                const float* __restrict__ cob, float* __restrict__ blk) {
    int i = blockIdx.x * blockDim.x + threadIdx.x;
    if (i >= E_EDGES * I_EDGE_D) return;
    const int e = i / I_EDGE_D, xy = i % I_EDGE_D;
    const float* ir = irr + (size_t)e * I_EDGE_D;
    const float* cb = cob + (size_t)et[e] * I_EDGE_D * I_EDGE_D + xy;
    float acc = 0.f;
#pragma unroll 13
    for (int k = 0; k < I_EDGE_D; ++k) acc = fmaf(ir[k], cb[(size_t)k * I_EDGE_D], acc);
    blk[i] = acc;
}

// ---------------- node irreps ----------------
__global__ void node_irr_kernel(const float* __restrict__ nh, const int* __restrict__ nt,
                                const float* __restrict__ W_node, float* __restrict__ irr) {
    int i = blockIdx.x * blockDim.x + threadIdx.x;
    if (i >= N_NODES * I_NODE_D) return;
    const int n = i / I_NODE_D, ix = i % I_NODE_D;
    const float* hr = nh + (size_t)n * F_DIM;
    const float* w = W_node + (size_t)nt[n] * F_DIM * I_NODE_D + ix;
    float acc = 0.f;
#pragma unroll 8
    for (int f = 0; f < F_DIM; ++f) acc = fmaf(hr[f], w[(size_t)f * I_NODE_D], acc);
    irr[i] = acc;
}

// ---------------- node blocks + diagonal scatter ----------------
__global__ void node_blk_scatter(const float* __restrict__ irr, const int* __restrict__ nt,
                                 const float* __restrict__ cob, float* __restrict__ M) {
    int i = blockIdx.x * blockDim.x + threadIdx.x;
    if (i >= N_NODES * I_EDGE_D) return;
    const int n = i / I_EDGE_D, xy = i % I_EDGE_D;
    const int x = xy / D_DIM, y = xy % D_DIM;
    const float* ir = irr + (size_t)n * I_NODE_D;
    const float* cb = cob + (size_t)nt[n] * I_NODE_D * I_EDGE_D + xy;
    float acc = 0.f;
#pragma unroll 13
    for (int k = 0; k < I_NODE_D; ++k) acc = fmaf(ir[k], cb[(size_t)k * I_EDGE_D], acc);
    atomicAdd(&M[(size_t)(n * D_DIM + x) * ND + n * D_DIM + y], acc);
}

// ---------------- edge block symmetrize + scatter ----------------
__global__ void edge_scatter(const float* __restrict__ bij, const float* __restrict__ bji,
                             const int* __restrict__ src, const int* __restrict__ dst,
                             float* __restrict__ M) {
    int i = blockIdx.x * blockDim.x + threadIdx.x;
    if (i >= E_EDGES * I_EDGE_D) return;
    const int e = i / I_EDGE_D, xy = i % I_EDGE_D;
    const int x = xy / D_DIM, y = xy % D_DIM;
    const float* pij = bij + (size_t)e * I_EDGE_D;
    const float* pji = bji + (size_t)e * I_EDGE_D;
    const float v1 = 0.5f * (pij[x * D_DIM + y] + pji[y * D_DIM + x]);  // eb[x][y]
    const float v2 = 0.5f * (pij[y * D_DIM + x] + pji[x * D_DIM + y]);  // eb[y][x]
    const int s = src[e], d = dst[e];
    atomicAdd(&M[(size_t)(s * D_DIM + x) * ND + d * D_DIM + y], v1);
    atomicAdd(&M[(size_t)(d * D_DIM + x) * ND + s * D_DIM + y], v2);
}

// ---------------- launcher ----------------
static inline int cdiv(int a, int b) { return (a + b - 1) / b; }

extern "C" void kernel_launch(void* const* d_in, const int* in_sizes, int n_in,
                              void* d_out, int out_size, void* d_ws, size_t ws_size,
                              hipStream_t stream) {
    const float* node_feats = (const float*)d_in[0];
    const float* node_attrs = (const float*)d_in[1];
    const float* edge_feats = (const float*)d_in[2];
    const float* edge_attrs = (const float*)d_in[3];
    const int*   edge_index = (const int*)d_in[4];
    const int*   node_types = (const int*)d_in[5];
    const int*   edge_types = (const int*)d_in[6];
    const float* W_msg  = (const float*)d_in[7];
    const float* W_attr = (const float*)d_in[8];
    const float* W_em   = (const float*)d_in[9];
    const float* W_proj = (const float*)d_in[10];
    const float* W_node = (const float*)d_in[11];
    const float* W_edge = (const float*)d_in[12];
    const float* cob_node = (const float*)d_in[13];
    const float* cob_edge = (const float*)d_in[14];
    float* M  = (float*)d_out;
    float* ws = (float*)d_ws;
    const int* src = edge_index;
    const int* dst = edge_index + E_EDGES;

    // ---- workspace layout (float units; every offset is a multiple of 32 floats = 128B) ----
    float* w_m       = ws;                        // 128000
    float* w_agg     = w_m      + 128000;         // 128000
    float* w_nh      = w_agg    + 128000;         // 128000
    float* w_nirr    = w_nh     + 128000;         // 96000 (91000 used)
    float* w_hij     = w_nirr   + 96000;          // 1,280,000
    float* w_hji     = w_hij    + 1280000;        // 1,280,000
    float* w_irr_ij  = w_hji    + 1280000;        // 3,380,000
    float* w_irr_ji  = w_irr_ij + 3380000;        // 3,380,000
    float* w_blk_ij  = w_irr_ji + 3380000;        // 3,380,000
    float* w_blk_ji  = w_blk_ij + 3380000;        // 3,380,000
    float* bf_base   = w_blk_ji + 3380000;
    // bf16 regions (sizes in float units = bf16count/2)
    __bf16* w_nf_bf   = (__bf16*)(bf_base);             // 128000 bf16 -> 64000 f
    __bf16* w_nh_bf   = (__bf16*)(bf_base + 64000);     // 128000 bf16
    __bf16* w_ef_bf   = (__bf16*)(bf_base + 128000);    // 320000 bf16 -> 160000 f
    __bf16* w_ea_bf   = (__bf16*)(bf_base + 288000);    // 320000 bf16
    __bf16* w_emsg_bf = (__bf16*)(bf_base + 448000);    // 2,560,000 bf16 -> 1,280,000 f
    __bf16* wt_msg    = (__bf16*)(bf_base + 1728000);   // 16384 bf16 -> 8192 f
    __bf16* wt_em     = (__bf16*)(bf_base + 1736192);   // 36864 bf16 -> 18432 f
    __bf16* wt_proj   = (__bf16*)(bf_base + 1754624);   // 24576 bf16 -> 12288 f

    // 1) zero output (676 MB, nontemporal) + zero agg
    {
        const int n4 = (ND * ND) / 4;
        zero_nt_f4<<<cdiv(n4, 256), 256, 0, stream>>>((v4f*)M, n4);
        zero_f4<<<cdiv(128000 / 4, 256), 256, 0, stream>>>((v4f*)w_agg, 128000 / 4);
    }
    // 2) precision staging: bf16 copies + transposed bf16 weights
    conv_bf16<<<cdiv(N_NODES * F_DIM, 256), 256, 0, stream>>>(node_feats, w_nf_bf, N_NODES * F_DIM);
    conv_bf16<<<cdiv(E_EDGES * EA_DIM, 256), 256, 0, stream>>>(edge_feats, w_ef_bf, E_EDGES * EA_DIM);
    conv_bf16<<<cdiv(E_EDGES * EA_DIM, 256), 256, 0, stream>>>(edge_attrs, w_ea_bf, E_EDGES * EA_DIM);
    transpose_bf16<<<cdiv(F_DIM * F_DIM, 256), 256, 0, stream>>>(W_msg, wt_msg, F_DIM, F_DIM);
    transpose_bf16<<<cdiv(FIN_EM * H_DIM, 256), 256, 0, stream>>>(W_em, wt_em, FIN_EM, H_DIM);
    transpose_bf16<<<cdiv(FIN_PROJ * K_DIM, 256), 256, 0, stream>>>(W_proj, wt_proj, FIN_PROJ, K_DIM);
    // 3) m = node_feats @ W_msg  (WMMA)
    gemm_node_msg<<<cdiv(63, 8), 256, 0, stream>>>(w_nf_bf, wt_msg, w_m);
    // 4) symmetric aggregation
    aggregate_kernel<<<cdiv(E_EDGES * F_DIM, 256), 256, 0, stream>>>(w_m, src, dst, w_agg);
    // 5) node_h (f32) + bf16 copy for GEMM A-operands
    node_h_kernel<<<cdiv(N_NODES * F_DIM, 256), 256, 0, stream>>>(node_feats, w_agg, node_attrs,
                                                                  W_attr, w_nh);
    conv_bf16<<<cdiv(N_NODES * F_DIM, 256), 256, 0, stream>>>(w_nh, w_nh_bf, N_NODES * F_DIM);
    // 6) edge messages (WMMA, fused gather + tanh, bf16 out)
    gemm_edge_msg<<<cdiv(E_EDGES / 16, 8), 256, 0, stream>>>(w_nh_bf, w_ef_bf, w_ea_bf,
                                                             src, dst, wt_em, w_emsg_bf);
    // 7) projections, both directions (WMMA, fused gather)
    gemm_proj<<<cdiv(E_EDGES / 16, 8), 256, 0, stream>>>(w_emsg_bf, w_nh_bf, src, dst, wt_proj, w_hij);
    gemm_proj<<<cdiv(E_EDGES / 16, 8), 256, 0, stream>>>(w_emsg_bf, w_nh_bf, dst, src, wt_proj, w_hji);
    // 8) per-edge irreps + change of basis
    {
        const int nt = E_EDGES * I_EDGE_D;
        edge_irr_kernel<<<cdiv(nt, 256), 256, 0, stream>>>(w_hij, edge_types, W_edge, w_irr_ij);
        edge_irr_kernel<<<cdiv(nt, 256), 256, 0, stream>>>(w_hji, edge_types, W_edge, w_irr_ji);
        edge_cob_kernel<<<cdiv(nt, 256), 256, 0, stream>>>(w_irr_ij, edge_types, cob_edge, w_blk_ij);
        edge_cob_kernel<<<cdiv(nt, 256), 256, 0, stream>>>(w_irr_ji, edge_types, cob_edge, w_blk_ji);
    }
    // 9) node self blocks -> diagonal
    node_irr_kernel<<<cdiv(N_NODES * I_NODE_D, 256), 256, 0, stream>>>(w_nh, node_types, W_node, w_nirr);
    node_blk_scatter<<<cdiv(N_NODES * I_EDGE_D, 256), 256, 0, stream>>>(w_nirr, node_types, cob_node, M);
    // 10) symmetrized edge blocks -> off-diagonal
    edge_scatter<<<cdiv(E_EDGES * I_EDGE_D, 256), 256, 0, stream>>>(w_blk_ij, w_blk_ji, src, dst, M);
}
